// DecoderLayer_62328565399946
// MI455X (gfx1250) — matrix operand order repaired and checked by reading, not compiled, not run on hardware
//
#include <hip/hip_runtime.h>
#include <math.h>

// ---------------------------------------------------------------------------
// CDNA5 (gfx1250) decoder layer: all GEMMs via v_wmma_f32_16x16x32_f16
// fp32 global data, converted to f16 in LDS, fp32 accumulate.
// Double-buffered LDS tiles: next tile staged through registers while the
// current tile is consumed by WMMA (one workgroup barrier per K-tile).
// ---------------------------------------------------------------------------

typedef __attribute__((ext_vector_type(16))) _Float16 v16h;
typedef __attribute__((ext_vector_type(8)))  _Float16 v8h;
typedef __attribute__((ext_vector_type(8)))  float    v8f;

union U8f  { v8f  v; float    f[8]; };
union U16h { v16h v; v8h      h[2]; };

// ---------------------------------------------------------------------------
// Generic batched GEMM:  D = alpha * (A @ B) [+ bias] [+ maskCoef*mask] [+ resid] [relu]
//   A: fp32 [M,K], row ld = lda, batch offset zb*sAb + zh*sAh  (z = zb*Hn + zh)
//   B: fp32; BT==1 -> stored [N,K] (ld=ldb, direct copy to LDS-Bt)
//            BT==0 -> stored [K,N] (ld=ldb, transposed into LDS-Bt)
//   D: fp32 [M,N], ld = ldd
// Tile sizes must divide M,N,K exactly (true for every launch below).
// ---------------------------------------------------------------------------
template<int BM, int BN, int BK, int WROWS, int WCOLS>
__global__ __launch_bounds__(256)
void gemm_wmma(const float* __restrict__ A,  long long sAb, long long sAh, int lda,
               const float* __restrict__ Bm, long long sBb, long long sBh, int ldb, int BT,
               float* __restrict__ D,        long long sDb, long long sDh, int ldd,
               const float* __restrict__ bias,
               const float* __restrict__ resid, int ldres,
               const float* __restrict__ mask, int maskMode, int ldmask, float maskCoef,
               int K, int Hn, float alpha, int relu)
{
    constexpr int LDS_A = BK + 8;           // f16 units; conflict-free row stride
    constexpr int LDS_B = BK + 8;
    constexpr int WTM = BM / WROWS, WTN = BN / WCOLS;
    constexpr int FM = WTM / 16, FN = WTN / 16, KS = BK / 32;
    constexpr int NA = (BM * BK / 4) / 256; // float4 loads per thread (A tile)
    constexpr int NB = (BN * BK / 4) / 256; // float4 loads per thread (B tile)

    __shared__ __align__(16) _Float16 sA[2][BM * LDS_A];
    __shared__ __align__(16) _Float16 sB[2][BN * LDS_B];   // stored as Bt[N][K]

    const int zb = blockIdx.z / Hn, zh = blockIdx.z % Hn;
    const float* Ab = A  + zb * sAb + zh * sAh + (long long)blockIdx.y * BM * lda;
    const float* Bb = Bm + zb * sBb + zh * sBh;

    const int t = threadIdx.x, lane = t & 31, wv = t >> 5;
    const int wr = wv / WCOLS, wc = wv % WCOLS;
    const int mBase = wr * WTM, nBase = wc * WTN;
    const int r16 = lane & 15, hi = lane >> 4;

    float4 ra[NA], rb[NB];                  // register staging for next tile

    auto loadA = [&](int kt) {
#pragma unroll
        for (int it = 0; it < NA; ++it) {
            int idx = t + it * 256;
            int row = idx / (BK / 4);
            int kq  = (idx % (BK / 4)) * 4;
            ra[it] = *(const float4*)(Ab + (long long)row * lda + kt + kq);
        }
    };
    auto storeA = [&](int buf) {
#pragma unroll
        for (int it = 0; it < NA; ++it) {
            int idx = t + it * 256;
            int row = idx / (BK / 4);
            int kq  = (idx % (BK / 4)) * 4;
            _Float16* d = &sA[buf][row * LDS_A + kq];
            d[0] = (_Float16)ra[it].x; d[1] = (_Float16)ra[it].y;
            d[2] = (_Float16)ra[it].z; d[3] = (_Float16)ra[it].w;
        }
    };
    auto loadB = [&](int kt) {
        if (BT) {            // B stored [N,K]: straight copy
#pragma unroll
            for (int it = 0; it < NB; ++it) {
                int idx = t + it * 256;
                int n  = idx / (BK / 4);
                int kq = (idx % (BK / 4)) * 4;
                rb[it] = *(const float4*)(Bb + ((long long)blockIdx.x * BN + n) * ldb + kt + kq);
            }
        } else {             // B stored [K,N]
#pragma unroll
            for (int it = 0; it < NB; ++it) {
                int idx = t + it * 256;
                int k  = idx / (BN / 4);
                int nq = (idx % (BN / 4)) * 4;
                rb[it] = *(const float4*)(Bb + (long long)(kt + k) * ldb
                                             + (long long)blockIdx.x * BN + nq);
            }
        }
    };
    auto storeB = [&](int buf) {
        if (BT) {
#pragma unroll
            for (int it = 0; it < NB; ++it) {
                int idx = t + it * 256;
                int n  = idx / (BK / 4);
                int kq = (idx % (BK / 4)) * 4;
                _Float16* d = &sB[buf][n * LDS_B + kq];
                d[0] = (_Float16)rb[it].x; d[1] = (_Float16)rb[it].y;
                d[2] = (_Float16)rb[it].z; d[3] = (_Float16)rb[it].w;
            }
        } else {             // transpose while storing
#pragma unroll
            for (int it = 0; it < NB; ++it) {
                int idx = t + it * 256;
                int k  = idx / (BN / 4);
                int nq = (idx % (BN / 4)) * 4;
                sB[buf][(nq + 0) * LDS_B + k] = (_Float16)rb[it].x;
                sB[buf][(nq + 1) * LDS_B + k] = (_Float16)rb[it].y;
                sB[buf][(nq + 2) * LDS_B + k] = (_Float16)rb[it].z;
                sB[buf][(nq + 3) * LDS_B + k] = (_Float16)rb[it].w;
            }
        }
    };

    v8f acc[FM][FN];
    {
        U8f z;
#pragma unroll
        for (int i = 0; i < 8; ++i) z.f[i] = 0.f;
#pragma unroll
        for (int mi = 0; mi < FM; ++mi)
#pragma unroll
            for (int ni = 0; ni < FN; ++ni) acc[mi][ni] = z.v;
    }

    // prologue: fill buffer 0
    loadA(0); loadB(0);
    storeA(0); storeB(0);
    __syncthreads();

    int cur = 0;
    for (int kt = 0; kt < K; kt += BK) {
        const bool hasNext = (kt + BK) < K;
        if (hasNext) { loadA(kt + BK); loadB(kt + BK); }   // global loads in flight

        // ---- WMMA on current buffer ----
        // ISA 16-bit A-frag and B-frag = two 16B chunks/lane (k 8h.., 16+8h..).
#pragma unroll
        for (int ks = 0; ks < KS; ++ks) {
            v16h af[FM], bf[FN];
#pragma unroll
            for (int mi = 0; mi < FM; ++mi) {
                const _Float16* p = &sA[cur][(mBase + mi * 16 + r16) * LDS_A + ks * 32 + hi * 8];
                U16h u; u.h[0] = *(const v8h*)p; u.h[1] = *(const v8h*)(p + 16);
                af[mi] = u.v;
            }
#pragma unroll
            for (int ni = 0; ni < FN; ++ni) {
                const _Float16* p = &sB[cur][(nBase + ni * 16 + r16) * LDS_B + ks * 32 + hi * 8];
                U16h u; u.h[0] = *(const v8h*)p; u.h[1] = *(const v8h*)(p + 16);
                bf[ni] = u.v;
            }
#pragma unroll
            for (int mi = 0; mi < FM; ++mi)
#pragma unroll
                for (int ni = 0; ni < FN; ++ni)
                    acc[mi][ni] = __builtin_amdgcn_wmma_f32_16x16x32_f16(
                        false, af[mi], false, bf[ni], (short)0, acc[mi][ni], false, false);
        }

        if (hasNext) { storeA(cur ^ 1); storeB(cur ^ 1); }
        __syncthreads();
        cur ^= 1;
    }

    // ---- epilogue: scale / bias / mask / residual / relu, fp32 store ----
    float* Db = D + zb * sDb + zh * sDh;
#pragma unroll
    for (int mi = 0; mi < FM; ++mi) {
#pragma unroll
        for (int ni = 0; ni < FN; ++ni) {
            U8f u; u.v = acc[mi][ni];
            const int gn = blockIdx.x * BN + nBase + ni * 16 + r16;
#pragma unroll
            for (int r = 0; r < 8; ++r) {
                const int gm = blockIdx.y * BM + mBase + mi * 16 + hi * 8 + r;
                float val = u.f[r] * alpha;
                if (bias)               val += bias[gn];
                if (maskMode == 1)      val += maskCoef * mask[(long long)gm * ldmask + gn];
                else if (maskMode == 2) val += maskCoef * mask[(long long)zb * ldmask + gn];
                if (resid)              val += resid[(long long)gm * ldres + gn];
                if (relu)               val  = fmaxf(val, 0.f);
                Db[(long long)gm * ldd + gn] = val;
            }
        }
    }
}

// ---------------------------------------------------------------------------
// wave32 shuffle reductions
// ---------------------------------------------------------------------------
__device__ inline float wred_max(float v) {
#pragma unroll
    for (int o = 16; o; o >>= 1) v = fmaxf(v, __shfl_xor(v, o));
    return v;
}
__device__ inline float wred_sum(float v) {
#pragma unroll
    for (int o = 16; o; o >>= 1) v += __shfl_xor(v, o);
    return v;
}

// In-place row softmax; rows = gridDim.x, cols <= 1536, 256 threads.
__global__ __launch_bounds__(256)
void softmax_rows(float* __restrict__ p, int cols)
{
    __shared__ float red[8];
    __shared__ float bval;
    float* r = p + (long long)blockIdx.x * cols;
    const int t = threadIdx.x, lane = t & 31, wv = t >> 5;

    float loc[8];
    int n = 0;
    float mx = -3.4e38f;
    for (int c = t; c < cols; c += 256) { float x = r[c]; loc[n++] = x; mx = fmaxf(mx, x); }

    mx = wred_max(mx);
    if (lane == 0) red[wv] = mx;
    __syncthreads();
    if (t < 32) { float v = (lane < 8) ? red[lane] : -3.4e38f; v = wred_max(v); if (lane == 0) bval = v; }
    __syncthreads();
    mx = bval;

    float s = 0.f;
    for (int i = 0; i < n; ++i) { loc[i] = __expf(loc[i] - mx); s += loc[i]; }
    s = wred_sum(s);
    __syncthreads();
    if (lane == 0) red[wv] = s;
    __syncthreads();
    if (t < 32) { float v = (lane < 8) ? red[lane] : 0.f; v = wred_sum(v); if (lane == 0) bval = v; }
    __syncthreads();

    const float inv = 1.f / bval;
    n = 0;
    for (int c = t; c < cols; c += 256) r[c] = loc[n++] * inv;
}

// LayerNorm over rows (cols = 512 here): out = g*(x-mu)*rsqrt(var+eps)+b
__global__ __launch_bounds__(256)
void ln_rows(const float* __restrict__ in, const float* __restrict__ g,
             const float* __restrict__ bb, float* __restrict__ out, int cols)
{
    __shared__ float red[8];
    __shared__ float bval;
    const float* r = in + (long long)blockIdx.x * cols;
    float* o = out + (long long)blockIdx.x * cols;
    const int t = threadIdx.x, lane = t & 31, wv = t >> 5;

    float loc[4];
    int n = 0;
    float s = 0.f;
    for (int c = t; c < cols; c += 256) { float x = r[c]; loc[n++] = x; s += x; }
    s = wred_sum(s);
    if (lane == 0) red[wv] = s;
    __syncthreads();
    if (t < 32) { float v = (lane < 8) ? red[lane] : 0.f; v = wred_sum(v); if (lane == 0) bval = v; }
    __syncthreads();
    const float mu = bval / (float)cols;

    float vs = 0.f;
    for (int i = 0; i < n; ++i) { float d = loc[i] - mu; vs += d * d; }
    vs = wred_sum(vs);
    __syncthreads();
    if (lane == 0) red[wv] = vs;
    __syncthreads();
    if (t < 32) { float v = (lane < 8) ? red[lane] : 0.f; v = wred_sum(v); if (lane == 0) bval = v; }
    __syncthreads();
    const float inv = rsqrtf(bval / (float)cols + 1e-6f);

    n = 0;
    for (int c = t; c < cols; c += 256) { o[c] = g[c] * (loc[n++] - mu) * inv + bb[c]; }
}

// ---------------------------------------------------------------------------
extern "C" void kernel_launch(void* const* d_in, const int* in_sizes, int n_in,
                              void* d_out, int out_size, void* d_ws, size_t ws_size,
                              hipStream_t stream)
{
    (void)in_sizes; (void)n_in; (void)out_size;
    constexpr int  B = 4, SQ = 1024, SK = 1536, E = 512, H = 8, HD = 64, FF = 2048;
    constexpr long long LSQ = SQ, LSK = SK, LE = E;

    const float* x        = (const float*)d_in[0];
    const float* enc      = (const float*)d_in[1];
    const float* la_mask  = (const float*)d_in[2];
    const float* pad_mask = (const float*)d_in[3];
    const float* m1_wq = (const float*)d_in[4],  *m1_bq = (const float*)d_in[5];
    const float* m1_wk = (const float*)d_in[6],  *m1_bk = (const float*)d_in[7];
    const float* m1_wv = (const float*)d_in[8],  *m1_bv = (const float*)d_in[9];
    const float* m1_wo = (const float*)d_in[10], *m1_bo = (const float*)d_in[11];
    const float* m2_wq = (const float*)d_in[12], *m2_bq = (const float*)d_in[13];
    const float* m2_wk = (const float*)d_in[14], *m2_bk = (const float*)d_in[15];
    const float* m2_wv = (const float*)d_in[16], *m2_bv = (const float*)d_in[17];
    const float* m2_wo = (const float*)d_in[18], *m2_bo = (const float*)d_in[19];
    const float* ffn_w1 = (const float*)d_in[20], *ffn_b1 = (const float*)d_in[21];
    const float* ffn_w2 = (const float*)d_in[22], *ffn_b2 = (const float*)d_in[23];
    const float* ln1_g = (const float*)d_in[24], *ln1_b = (const float*)d_in[25];
    const float* ln2_g = (const float*)d_in[26], *ln2_b = (const float*)d_in[27];
    const float* ln3_g = (const float*)d_in[28], *ln3_b = (const float*)d_in[29];

    float* out3 = (float*)d_out;
    float* aw1  = out3 + (long long)B * SQ * E;          // [B,H,SQ,SQ]
    float* aw2  = aw1  + (long long)B * H * SQ * SQ;     // [B,H,SQ,SK]

    if (ws_size < (size_t)16777216 * sizeof(float)) return;   // 64 MB carve below
    float* W    = (float*)d_ws;
    float* out1 = W;                    // 2M
    float* out2 = out1 + 2097152;       // 2M
    float* sum  = out2 + 2097152;       // 2M
    float* cat  = sum  + 2097152;       // 2M
    float* qb   = cat  + 2097152;       // 2M
    float* kb   = qb   + 2097152;       // 3M
    float* vb   = kb   + 3145728;       // 3M
    float* ffnh = qb;                   // 8M, reuses q/k/v region after MHA2

    const dim3 blk(256);
    const float a1 = 1.f / sqrtf((float)SQ);     // reference scales by sqrt(Sk)
    const float a2 = 1.f / sqrtf((float)SK);

    // ---------------- MHA1 (self-attention, causal) ----------------
    // per-head projections: [B*SQ*H, 64] @ [64,64] + bias
    gemm_wmma<128,64,64,8,1><<<dim3(1,256,1),blk,0,stream>>>(
        x,0,0,HD, m1_wq,0,0,HD,0, qb,0,0,HD, m1_bq, nullptr,0, nullptr,0,0,0.f, HD,1,1.f,0);
    gemm_wmma<128,64,64,8,1><<<dim3(1,256,1),blk,0,stream>>>(
        x,0,0,HD, m1_wk,0,0,HD,0, kb,0,0,HD, m1_bk, nullptr,0, nullptr,0,0,0.f, HD,1,1.f,0);
    gemm_wmma<128,64,64,8,1><<<dim3(1,256,1),blk,0,stream>>>(
        x,0,0,HD, m1_wv,0,0,HD,0, vb,0,0,HD, m1_bv, nullptr,0, nullptr,0,0,0.f, HD,1,1.f,0);
    // logits1 = Q K^T / sqrt(SQ) - 1e9*causal_mask -> aw1 (K-matrix rows are Bt directly)
    gemm_wmma<128,128,32,4,2><<<dim3(8,8,32),blk,0,stream>>>(
        qb, LSQ*LE, HD, E,  kb, LSQ*LE, HD, E, 1,
        aw1, 8LL*LSQ*LSQ, LSQ*LSQ, SQ,
        nullptr, nullptr,0, la_mask,1,SQ,-1e9f, HD,8,a1,0);
    softmax_rows<<<B*H*SQ,blk,0,stream>>>(aw1, SQ);
    // AV1 -> cat [B,SQ,E] (heads interleaved at stride HD)
    gemm_wmma<128,64,64,8,1><<<dim3(1,8,32),blk,0,stream>>>(
        aw1, 8LL*LSQ*LSQ, LSQ*LSQ, SQ,  vb, LSQ*LE, HD, E, 0,
        cat, LSQ*LE, HD, E, nullptr, nullptr,0, nullptr,0,0,0.f, SQ,8,1.f,0);
    // out-proj + residual(x) -> sum; LN1 -> out1
    gemm_wmma<128,128,32,4,2><<<dim3(4,32,1),blk,0,stream>>>(
        cat,0,0,E, m1_wo,0,0,E,0, sum,0,0,E, m1_bo, x,E, nullptr,0,0,0.f, E,1,1.f,0);
    ln_rows<<<B*SQ,blk,0,stream>>>(sum, ln1_g, ln1_b, out1, E);

    // ---------------- MHA2 (cross-attention) ----------------
    gemm_wmma<128,64,64,8,1><<<dim3(1,256,1),blk,0,stream>>>(
        out1,0,0,HD, m2_wq,0,0,HD,0, qb,0,0,HD, m2_bq, nullptr,0, nullptr,0,0,0.f, HD,1,1.f,0);
    gemm_wmma<128,64,64,8,1><<<dim3(1,384,1),blk,0,stream>>>(
        enc,0,0,HD, m2_wk,0,0,HD,0, kb,0,0,HD, m2_bk, nullptr,0, nullptr,0,0,0.f, HD,1,1.f,0);
    gemm_wmma<128,64,64,8,1><<<dim3(1,384,1),blk,0,stream>>>(
        enc,0,0,HD, m2_wv,0,0,HD,0, vb,0,0,HD, m2_bv, nullptr,0, nullptr,0,0,0.f, HD,1,1.f,0);
    gemm_wmma<128,128,32,4,2><<<dim3(12,8,32),blk,0,stream>>>(
        qb, LSQ*LE, HD, E,  kb, LSK*LE, HD, E, 1,
        aw2, 8LL*LSQ*LSK, LSQ*LSK, SK,
        nullptr, nullptr,0, pad_mask,2,SK,-1e9f, HD,8,a2,0);
    softmax_rows<<<B*H*SQ,blk,0,stream>>>(aw2, SK);
    gemm_wmma<128,64,64,8,1><<<dim3(1,8,32),blk,0,stream>>>(
        aw2, 8LL*LSQ*LSK, LSQ*LSK, SK,  vb, LSK*LE, HD, E, 0,
        cat, LSQ*LE, HD, E, nullptr, nullptr,0, nullptr,0,0,0.f, SK,8,1.f,0);
    gemm_wmma<128,128,32,4,2><<<dim3(4,32,1),blk,0,stream>>>(
        cat,0,0,E, m2_wo,0,0,E,0, sum,0,0,E, m2_bo, out1,E, nullptr,0,0,0.f, E,1,1.f,0);
    ln_rows<<<B*SQ,blk,0,stream>>>(sum, ln2_g, ln2_b, out2, E);

    // ---------------- FFN ----------------
    gemm_wmma<128,128,32,4,2><<<dim3(16,32,1),blk,0,stream>>>(
        out2,0,0,E, ffn_w1,0,0,FF,0, ffnh,0,0,FF, ffn_b1, nullptr,0, nullptr,0,0,0.f, E,1,1.f,1);
    gemm_wmma<128,128,32,4,2><<<dim3(4,32,1),blk,0,stream>>>(
        ffnh,0,0,FF, ffn_w2,0,0,E,0, sum,0,0,E, ffn_b2, out2,E, nullptr,0,0,0.f, FF,1,1.f,0);
    ln_rows<<<B*SQ,blk,0,stream>>>(sum, ln3_g, ln3_b, out3, E);
}
